// MultiHeadAttention_48095043781107
// MI455X (gfx1250) — matrix operand order repaired and checked by reading, not compiled, not run on hardware
//
#include <hip/hip_runtime.h>
#include <hip/hip_bf16.h>
#include <math.h>

// ---------------------------------------------------------------------------
// MHA forward for gfx1250 (MI455X): bf16 WMMA + fp32 accumulate.
//  - GEMMs stage the shared A-strip in LDS via GLOBAL_LOAD_ASYNC_TO_LDS_B128
//  - attn (536 MB) streamed to HBM with nontemporal stores (bypass L2 reuse)
// Outputs: out [2,2048,1024] fp32, then attn [2,16,2048,2048] fp32.
// ---------------------------------------------------------------------------

typedef __bf16 bf16;
typedef __attribute__((ext_vector_type(16))) __bf16 v16bf;
typedef __attribute__((ext_vector_type(8)))  __bf16 v8bf;
typedef __attribute__((ext_vector_type(8)))  float  v8f;

union AFrag { v16bf v; v8bf h[2]; };

#define BATCH   2
#define SEQ     2048
#define DMODEL  1024
#define HEADS   16
#define DEPTH   64
#define MROWS   (BATCH * SEQ)              // 4096
#define OUT_OFF ((size_t)MROWS * DMODEL)   // float offset of attn in d_out

// LDS A-strip row stride (bf16 elems): 1024 + 8 -> 516 dwords/row -> 4-bank
// shift per row, so 16 lanes reading 16 rows at one column are conflict-free.
#define ASTRIDE 1032

static __device__ __forceinline__ v8f wmma_bf16(const v16bf& a, const v16bf& b, const v8f& c) {
    return __builtin_amdgcn_wmma_f32_16x16x32_bf16(false, a, false, b, (short)0, c, false, false);
}

// Cooperative async copy: 16 rows x 2048B of X into padded LDS strip.
// 128 threads x 16 chunks of 16B, ASYNCcnt-tracked, then barrier.
static __device__ __forceinline__ void stage_A_async(const bf16* __restrict__ Xrow0,
                                                     bf16* __restrict__ As) {
    const int t    = threadIdx.x;
    const int row  = t & 15;
    const int cb   = t >> 4;                       // 0..7
#if defined(__HIP_DEVICE_COMPILE__)
    const uint32_t lds_row = (uint32_t)(uintptr_t)(As + row * ASTRIDE);
    const uint64_t g_row   = (uint64_t)(uintptr_t)(Xrow0 + (size_t)row * DMODEL);
#pragma unroll
    for (int i = 0; i < 16; ++i) {
        const int c = cb + i * 8;                  // 0..127 16-byte chunks
        const uint32_t l = lds_row + c * 16;
        const uint64_t g = g_row + c * 16;
        asm volatile("global_load_async_to_lds_b128 %0, %1, off"
                     :: "v"(l), "v"(g) : "memory");
    }
    asm volatile("s_wait_asynccnt 0x0" ::: "memory");
#else
    for (int i = 0; i < 16; ++i) {
        const int c = cb + i * 8;
        ((float4*)(As + row * ASTRIDE))[c] = ((const float4*)(Xrow0 + (size_t)row * DMODEL))[c];
    }
#endif
    __syncthreads();
}

// ---------------------------------------------------------------- casts ----
__global__ void cast_to_bf16(const float* __restrict__ in, bf16* __restrict__ out, int n) {
    int i = blockIdx.x * blockDim.x + threadIdx.x;
    if (i < n) out[i] = (bf16)in[i];
}

// Wt[n][k] = W[k][n]  (1024 x 1024), so WMMA B-fragments load contiguously.
__global__ void cast_transpose_bf16(const float* __restrict__ W, bf16* __restrict__ Wt) {
    int i = blockIdx.x * blockDim.x + threadIdx.x;   // over 1024*1024
    int r = i >> 10;          // output row (n)
    int c = i & 1023;         // output col (k)
    Wt[(size_t)r * DMODEL + c] = (bf16)W[(size_t)c * DMODEL + r];
}

// ------------------------------------------------- projection GEMM (bf16) --
// C = X[4096x1024] * Wt^T + bias.  Each wave: 16(M) x 64(N); block = 4 waves.
// mode 0: out bf16 [B,H,S,64]   mode 1: out bf16 [B,H,64,S]  (V^T)
__global__ void gemm_proj_bf16(const bf16* __restrict__ X, const bf16* __restrict__ Wt,
                               const float* __restrict__ bias, bf16* __restrict__ out, int mode) {
    __shared__ __align__(16) bf16 As[16 * ASTRIDE];
    const int lane = threadIdx.x & 31;
    const int wv   = threadIdx.x >> 5;
    const int half = lane >> 4;
    const int l16  = lane & 15;
    const int m0   = blockIdx.x * 16;
    const int n0   = blockIdx.y * 256 + wv * 64;

    stage_A_async(X + (size_t)m0 * DMODEL, As);

    v8f acc[4] = {};
    const bf16* arow = As + l16 * ASTRIDE;
    const bf16* brow0 = Wt + (size_t)(n0 + l16) * DMODEL;

    for (int k0 = 0; k0 < DMODEL; k0 += 32) {
        __builtin_prefetch(brow0 + k0 + 512, 0, 1);          // global_prefetch_b8
        AFrag a;
        a.h[0] = *(const v8bf*)(arow + k0 + half * 16);      // ds_load_b128
        a.h[1] = *(const v8bf*)(arow + k0 + half * 16 + 8);
        const int bseg = k0 + half * 16;
#pragma unroll
        for (int j = 0; j < 4; ++j) {
            v16bf b = *(const v16bf*)(Wt + (size_t)(n0 + j * 16 + l16) * DMODEL + bseg);
            acc[j] = wmma_bf16(a.v, b, acc[j]);
        }
    }

#pragma unroll
    for (int j = 0; j < 4; ++j) {
        const int n  = n0 + j * 16 + l16;
        const float bv = bias[n];
        const int h = n >> 6, d = n & 63;
#pragma unroll
        for (int r = 0; r < 8; ++r) {
            const int m = m0 + r + half * 8;          // D layout: lanes16-31 -> M+8
            const int b = m >> 11, s = m & 2047;
            const float val = acc[j][r] + bv;
            if (mode == 0)
                out[((size_t)(b * HEADS + h) * SEQ + s) * DEPTH + d] = (bf16)val;
            else
                out[((size_t)(b * HEADS + h) * DEPTH + d) * SEQ + s] = (bf16)val;
        }
    }
}

// -------------------------------------------------- fused attention core ---
// One block = 4 waves = one (b,h,16-query) tile. Full 16x2048 fp32 logit row
// in LDS (128 KB); softmax in-place; attn streamed to HBM once (nontemporal);
// ctx = attn @ V via WMMA reading V^T.
__global__ void attn_fused(const bf16* __restrict__ qh, const bf16* __restrict__ kh,
                           const bf16* __restrict__ vt, bf16* __restrict__ ctx,
                           float* __restrict__ attn_out) {
    extern __shared__ float lg[];                     // [16][2048] fp32
    __shared__ float pm[16][8], ps[16][8], rowmax[16], rowsum[16];

    const int bh = blockIdx.x >> 7;                   // b*16 + h
    const int q0 = (blockIdx.x & 127) << 4;
    const int lane = threadIdx.x & 31;
    const int wv   = threadIdx.x >> 5;
    const int half = lane >> 4;
    const int l16  = lane & 15;

    const bf16* qb = qh + (size_t)bh * SEQ * DEPTH;
    const bf16* kb = kh + (size_t)bh * SEQ * DEPTH;
    const bf16* vb = vt + (size_t)bh * DEPTH * SEQ;
    float* ab = attn_out + (size_t)bh * SEQ * SEQ + (size_t)q0 * SEQ;

    // ---- phase 1: logits = scale * Q K^T; wave wv owns keys [wv*512, +512)
    const bf16* arow = qb + (size_t)(q0 + l16) * DEPTH;
    AFrag a_lo, a_hi;
    a_lo.h[0] = *(const v8bf*)(arow + half * 16);
    a_lo.h[1] = *(const v8bf*)(arow + half * 16 + 8);
    a_hi.h[0] = *(const v8bf*)(arow + 32 + half * 16);
    a_hi.h[1] = *(const v8bf*)(arow + 32 + half * 16 + 8);
    const float scale = 0.125f;                        // 1/sqrt(64)

    for (int t = 0; t < 32; ++t) {
        const int k0 = wv * 512 + t * 16;
        const bf16* krow = kb + (size_t)(k0 + l16) * DEPTH;
        v16bf b_lo = *(const v16bf*)(krow + half * 16);
        v16bf b_hi = *(const v16bf*)(krow + 32 + half * 16);
        v8f acc = {};
        acc = wmma_bf16(a_lo.v, b_lo, acc);
        acc = wmma_bf16(a_hi.v, b_hi, acc);
#pragma unroll
        for (int r = 0; r < 8; ++r) {
            const int m = r + half * 8;
            lg[m * SEQ + k0 + l16] = acc[r] * scale;
        }
    }
    __syncthreads();

    // ---- phase 2: softmax over 16 rows x 2048, 8 threads per row
    {
        const int tid  = threadIdx.x;
        const int row  = tid >> 3;
        const int part = tid & 7;
        float* seg = lg + row * SEQ + part * 256;
        float mx = -3.4e38f;
        for (int j = 0; j < 256; ++j) mx = fmaxf(mx, seg[j]);
        pm[row][part] = mx;
        __syncthreads();
        if (part == 0) {
            float m2 = pm[row][0];
#pragma unroll
            for (int j = 1; j < 8; ++j) m2 = fmaxf(m2, pm[row][j]);
            rowmax[row] = m2;
        }
        __syncthreads();
        mx = rowmax[row];
        float sum = 0.f;
        for (int j = 0; j < 256; ++j) {
            float e = __expf(seg[j] - mx);
            seg[j] = e;
            sum += e;
        }
        ps[row][part] = sum;
        __syncthreads();
        if (part == 0) {
            float s2 = 0.f;
#pragma unroll
            for (int j = 0; j < 8; ++j) s2 += ps[row][j];
            rowsum[row] = s2;
        }
        __syncthreads();
        const float inv = 1.0f / rowsum[row];
        float* og = ab + (size_t)row * SEQ + part * 256;
        for (int j = 0; j < 256; ++j) {
            float v = seg[j] * inv;
            seg[j] = v;
            __builtin_nontemporal_store(v, og + j);    // attn: write-once stream
        }
    }
    __syncthreads();

    // ---- phase 3: ctx(16 x 16-slice of d) = attn @ V; wave wv owns d range
    {
        const float* lrow = lg + l16 * SEQ;            // A row = lane%16
        const int dcol = wv * 16 + l16;                // B column (d index)
        v8f acc = {};
        for (int k0 = 0; k0 < SEQ; k0 += 32) {
            AFrag a;
#pragma unroll
            for (int j = 0; j < 8; ++j) {
                a.v[j]     = (bf16)lrow[k0 + half * 16 + j];
                a.v[8 + j] = (bf16)lrow[k0 + half * 16 + 8 + j];
            }
            v16bf b = *(const v16bf*)(vb + (size_t)dcol * SEQ + k0 + half * 16);
            acc = wmma_bf16(a.v, b, acc);
        }
        const int b = bh >> 4, h = bh & 15;
#pragma unroll
        for (int r = 0; r < 8; ++r) {
            const int m = r + half * 8;
            ctx[((size_t)b * SEQ + q0 + m) * DMODEL + h * DEPTH + wv * 16 + l16] = (bf16)acc[r];
        }
    }
}

// --------------------------------------------- output projection (fp32) ----
__global__ void gemm_out_f32(const bf16* __restrict__ X, const bf16* __restrict__ Wt,
                             const float* __restrict__ bias, float* __restrict__ out) {
    __shared__ __align__(16) bf16 As[16 * ASTRIDE];
    const int lane = threadIdx.x & 31;
    const int wv   = threadIdx.x >> 5;
    const int half = lane >> 4;
    const int l16  = lane & 15;
    const int m0   = blockIdx.x * 16;
    const int n0   = blockIdx.y * 256 + wv * 64;

    stage_A_async(X + (size_t)m0 * DMODEL, As);

    v8f acc[4] = {};
    const bf16* arow = As + l16 * ASTRIDE;
    const bf16* brow0 = Wt + (size_t)(n0 + l16) * DMODEL;
    for (int k0 = 0; k0 < DMODEL; k0 += 32) {
        __builtin_prefetch(brow0 + k0 + 512, 0, 1);
        AFrag a;
        a.h[0] = *(const v8bf*)(arow + k0 + half * 16);
        a.h[1] = *(const v8bf*)(arow + k0 + half * 16 + 8);
        const int bseg = k0 + half * 16;
#pragma unroll
        for (int j = 0; j < 4; ++j) {
            v16bf b = *(const v16bf*)(Wt + (size_t)(n0 + j * 16 + l16) * DMODEL + bseg);
            acc[j] = wmma_bf16(a.v, b, acc[j]);
        }
    }
#pragma unroll
    for (int j = 0; j < 4; ++j) {
        const int n = n0 + j * 16 + l16;
        const float bv = bias[n];
#pragma unroll
        for (int r = 0; r < 8; ++r) {
            const int m = m0 + r + half * 8;
            __builtin_nontemporal_store(acc[j][r] + bv, out + (size_t)m * DMODEL + n);
        }
    }
}

// ---------------------------------------------------------------------------
extern "C" void kernel_launch(void* const* d_in, const int* in_sizes, int n_in,
                              void* d_out, int out_size, void* d_ws, size_t ws_size,
                              hipStream_t stream) {
    const float* q  = (const float*)d_in[0];
    const float* k  = (const float*)d_in[1];
    const float* v  = (const float*)d_in[2];
    const float* wq = (const float*)d_in[3];
    const float* bq = (const float*)d_in[4];
    const float* wk = (const float*)d_in[5];
    const float* bk = (const float*)d_in[6];
    const float* wv = (const float*)d_in[7];
    const float* bv = (const float*)d_in[8];
    const float* wo = (const float*)d_in[9];
    const float* bo = (const float*)d_in[10];
    float* out = (float*)d_out;

    char* ws = (char*)d_ws;
    const size_t XE = (size_t)MROWS * DMODEL;            // 4M elements
    const size_t WE = (size_t)DMODEL * DMODEL;           // 1M elements
    bf16* qb  = (bf16*)(ws);                      // 8 MB
    bf16* kb  = (bf16*)(ws + 8u  * 1024 * 1024);
    bf16* vb  = (bf16*)(ws + 16u * 1024 * 1024);
    bf16* wqt = (bf16*)(ws + 24u * 1024 * 1024);  // 2 MB each
    bf16* wkt = (bf16*)(ws + 26u * 1024 * 1024);
    bf16* wvt = (bf16*)(ws + 28u * 1024 * 1024);
    bf16* wot = (bf16*)(ws + 30u * 1024 * 1024);
    bf16* qhh = (bf16*)(ws + 32u * 1024 * 1024);  // 8 MB each
    bf16* khh = (bf16*)(ws + 40u * 1024 * 1024);
    bf16* vtt = (bf16*)(ws + 48u * 1024 * 1024);  // V^T [B,H,64,S]
    bf16* ctx = (bf16*)(ws + 56u * 1024 * 1024);  // merged heads [B,S,1024]

    // 1) casts
    cast_to_bf16<<<(int)(XE / 256), 256, 0, stream>>>(q, qb, (int)XE);
    cast_to_bf16<<<(int)(XE / 256), 256, 0, stream>>>(k, kb, (int)XE);
    cast_to_bf16<<<(int)(XE / 256), 256, 0, stream>>>(v, vb, (int)XE);
    cast_transpose_bf16<<<(int)(WE / 256), 256, 0, stream>>>(wq, wqt);
    cast_transpose_bf16<<<(int)(WE / 256), 256, 0, stream>>>(wk, wkt);
    cast_transpose_bf16<<<(int)(WE / 256), 256, 0, stream>>>(wv, wvt);
    cast_transpose_bf16<<<(int)(WE / 256), 256, 0, stream>>>(wo, wot);

    // 2) projections: Q,K head-major; V transposed
    dim3 ggrid(MROWS / 16, DMODEL / 256);
    gemm_proj_bf16<<<ggrid, 128, 0, stream>>>(qb, wqt, bq, qhh, 0);
    gemm_proj_bf16<<<ggrid, 128, 0, stream>>>(kb, wkt, bk, khh, 0);
    gemm_proj_bf16<<<ggrid, 128, 0, stream>>>(vb, wvt, bv, vtt, 1);

    // 3) fused attention (128 KB dynamic LDS per block)
    (void)hipFuncSetAttribute((const void*)attn_fused,
                              hipFuncAttributeMaxDynamicSharedMemorySize, 131072);
    attn_fused<<<BATCH * HEADS * (SEQ / 16), 128, 131072, stream>>>(
        qhh, khh, vtt, ctx, out + OUT_OFF);

    // 4) output projection
    gemm_out_f32<<<ggrid, 128, 0, stream>>>(ctx, wot, bo, out);

    (void)in_sizes; (void)n_in; (void)out_size; (void)ws_size;
}
